// ResnetBlock_deform_9380208575102
// MI455X (gfx1250) — compile-verified, matrix operand order checked by reading, not run on hardware
//
#include <hip/hip_runtime.h>

// ---------------------------------------------------------------------------
// Problem constants (from reference): B=2, C=128, H=W=192, K=9
// ---------------------------------------------------------------------------
#define Bn 2
#define Cc 128
#define Hh 192
#define Ww 192
#define Kk 9
#define HW (Hh * Ww)
#define KD (Cc * Kk)      // 1152 = GEMM reduction dim (c*9 + tap)
#define NKC (KD / 32)     // 36 chunks of K=32 for wmma_f32_16x16x32_bf16

typedef __attribute__((ext_vector_type(16))) __bf16 v16bf;
typedef __attribute__((ext_vector_type(8)))  float  v8f;

// ---------------------------------------------------------------------------
// WMMA bf16 fragment layout (wave32, 16x16x32): lane l serves row/col l&15,
// khalf = l>>4.  Its 16 bf16 values (VGPR0..7, 2 per VGPR):
//   j<8  -> k = khalf*8 + j          (K 0..15 region)
//   j>=8 -> k = 16 + khalf*8 + (j-8) (K 16..31 region)
// ---------------------------------------------------------------------------
__device__ __forceinline__ int frag_swizzle(int kd, int p) {
  int kc = kd >> 5, kr = kd & 31, half, j;
  if (kr < 16) { half = kr >> 3; j = kr & 7; }
  else         { int k2 = kr - 16; half = k2 >> 3; j = 8 + (k2 & 7); }
  return (((kc * 16 + p) * 2 + half) << 4) + j;
}

// Pack main weights w[(o=128),(c=128),3,3] f32 -> bf16 B-fragments.
// packed index: (((kc*128 + n)*2 + khalf)*16 + j)
__global__ __launch_bounds__(256) void pack_w_kernel(const float* __restrict__ w,
                                                     __bf16* __restrict__ pw) {
  int i = blockIdx.x * 256 + threadIdx.x;           // < 36*128*2*16 = 147456
  int j    = i & 15;
  int half = (i >> 4) & 1;
  int n    = (i >> 5) & 127;
  int kc   = i >> 12;
  int kl   = (j < 8) ? (half * 8 + j) : (16 + half * 8 + (j - 8));
  int kd   = kc * 32 + kl;                           // 0..1151
  int c    = kd / Kk;
  int kk   = kd - c * Kk;
  pw[i] = (__bf16)w[((size_t)n * Cc + c) * 9 + kk];
}

// Pack offset-conv weights w_om[27,128,3,3] -> bf16 B-fragments, N padded
// 27 -> 32 (2 N-tiles).  packed index: (((kc*32 + n)*2 + khalf)*16 + j)
__global__ __launch_bounds__(256) void pack_wom_kernel(const float* __restrict__ w,
                                                       __bf16* __restrict__ pw) {
  int i = blockIdx.x * 256 + threadIdx.x;           // < 36*32*2*16 = 36864
  int j    = i & 15;
  int half = (i >> 4) & 1;
  int n    = (i >> 5) & 31;
  int kc   = i >> 10;
  int kl   = (j < 8) ? (half * 8 + j) : (16 + half * 8 + (j - 8));
  int kd   = kc * 32 + kl;
  int c    = kd / Kk;
  int kk   = kd - c * Kk;
  pw[i] = (n < 27) ? (__bf16)w[((size_t)n * Cc + c) * 9 + kk] : (__bf16)0.0f;
}

// ---------------------------------------------------------------------------
// Fully fused DCN layer: offset conv (WMMA) + bilinear sampling + einsum
// (WMMA).  One workgroup = 16 consecutive pixels x all 128 output channels.
// 8 waves.  The 36 KB LDS A-buffer is used twice: first for the im2col of
// `lo` (offset conv), then overwritten with the sampled/masked values.
// ---------------------------------------------------------------------------
__global__ __launch_bounds__(256) void dcn_fused_kernel(
    const float* __restrict__ src,    // tensor being sampled [B,C,H,W]
    const float* __restrict__ lo,     // offset-conv input    [B,C,H,W]
    const v16bf* __restrict__ pwom,   // packed offset-conv weights (N=32)
    const float* __restrict__ bom,    // offset-conv bias [27]
    const v16bf* __restrict__ pw,     // packed main weights (N=128)
    float* __restrict__ out) {        // [B,C,H,W] (pre-norm)
  __shared__ v16bf smA[NKC * 16 * 2];       // 36 KB A fragments (reused)
  __shared__ float smOff[27 * 16];          // offset-conv result per pixel
  __shared__ int   smOfs[16 * Kk * 4];      // bilinear corner offsets
  __shared__ float smWgt[16 * Kk * 4];      // corner weights (mask folded in)

  const int tid  = threadIdx.x;
  const int tile = blockIdx.x;              // B*H*(W/16) = 4608 tiles
  const int tilesPerRow = Ww / 16;          // 12
  const int row = tile / tilesPerRow;       // flattened (b, y)
  const int x0  = (tile % tilesPerRow) * 16;
  const int b   = row / Hh;
  const int y   = row % Hh;

  const int wave = tid >> 5;
  const int lane = tid & 31;
  const int ml   = lane & 15;               // M index (A/D) and N index (B)
  const int half = lane >> 4;

  __bf16* smAe = (__bf16*)smA;

  // ---- Phase 0a: im2col of lo (16 pixels x 1152 taps) into A fragments ----
  for (int e = tid; e < 16 * KD; e += 256) {
    int p  = e / KD;
    int kd = e - p * KD;                    // kd = c*9 + kk
    int c  = kd / Kk;
    int kk = kd - c * Kk;
    int yy = y + kk / 3 - 1;
    int xx = x0 + p + (kk % 3) - 1;
    float v = (yy >= 0 && yy < Hh && xx >= 0 && xx < Ww)
                  ? lo[(size_t)(b * Cc + c) * HW + yy * Ww + xx] : 0.0f;
    smAe[frag_swizzle(kd, p)] = (__bf16)v;
  }
  __syncthreads();

  // ---- Phase 0b: offset conv on WMMA (2 waves, N = 27 padded to 32) ----
  if (wave < 2) {                           // wave-uniform: EXEC stays full
    const int n = wave * 16 + ml;           // 0..31 (27..31 = padding)
    const v16bf* bb = pwom + ((size_t)n * 2 + half);
    v8f acc = {};
    for (int kc = 0; kc < NKC; ++kc) {
      v16bf a  = smA[(kc * 16 + ml) * 2 + half];
      v16bf bf = bb[(size_t)kc * 64];       // 32 n * 2 half frags per chunk
      acc = __builtin_amdgcn_wmma_f32_16x16x32_bf16(
          false, a, false, bf, (short)0, acc, false, false);
    }
    if (n < 27) {
      float bias = bom[n];
      #pragma unroll
      for (int r = 0; r < 8; ++r) smOff[n * 16 + (r + 8 * half)] = acc[r] + bias;
    }
  }
  __syncthreads();

  // ---- Phase 1: 144 bilinear corner descriptors (pixel p, tap kk) ----
  // Channel map: dy[k] = om[2k], dx[k] = om[2k+1], mask = sigmoid(om[18+k]).
  if (tid < 16 * Kk) {
    int p  = tid / Kk;
    int kk = tid - p * Kk;
    int xx = x0 + p;
    float dy = smOff[(2 * kk)     * 16 + p];
    float dx = smOff[(2 * kk + 1) * 16 + p];
    float mv = smOff[(18 + kk)    * 16 + p];
    mv = 1.0f / (1.0f + __expf(-mv));                // sigmoid(mask)
    float py = (float)y  + (float)(kk / 3 - 1) + dy;
    float px = (float)xx + (float)(kk % 3 - 1) + dx;
    float fy = floorf(py), fx = floorf(px);
    float ay = py - fy,    ax = px - fx;
    int y0 = (int)fy, x0i = (int)fx;
    int y1 = y0 + 1,  x1  = x0i + 1;
    bool vy0 = (y0  >= 0) && (y0  < Hh), vy1 = (y1 >= 0) && (y1 < Hh);
    bool vx0 = (x0i >= 0) && (x0i < Ww), vx1 = (x1 >= 0) && (x1 < Ww);
    int y0c = min(max(y0, 0), Hh - 1), y1c = min(max(y1, 0), Hh - 1);
    int x0c = min(max(x0i, 0), Ww - 1), x1c = min(max(x1, 0), Ww - 1);
    float w00 = (1.f - ay) * (1.f - ax) * mv, w01 = (1.f - ay) * ax * mv;
    float w10 = ay * (1.f - ax) * mv,         w11 = ay * ax * mv;
    smOfs[tid * 4 + 0] = y0c * Ww + x0c;  smWgt[tid * 4 + 0] = (vy0 && vx0) ? w00 : 0.f;
    smOfs[tid * 4 + 1] = y0c * Ww + x1c;  smWgt[tid * 4 + 1] = (vy0 && vx1) ? w01 : 0.f;
    smOfs[tid * 4 + 2] = y1c * Ww + x0c;  smWgt[tid * 4 + 2] = (vy1 && vx0) ? w10 : 0.f;
    smOfs[tid * 4 + 3] = y1c * Ww + x1c;  smWgt[tid * 4 + 3] = (vy1 && vx1) ? w11 : 0.f;
  }
  __syncthreads();

  // ---- Phase 2: sample 16 x 1152 values, overwrite A fragments ----
  for (int e = tid; e < 16 * KD; e += 256) {
    int p  = e / KD;
    int kd = e - p * KD;
    int c  = kd / Kk;
    int kk = kd - c * Kk;
    int pk = p * Kk + kk;
    const float* basep = src + (size_t)(b * Cc + c) * HW;
    float v = smWgt[pk * 4 + 0] * basep[smOfs[pk * 4 + 0]]
            + smWgt[pk * 4 + 1] * basep[smOfs[pk * 4 + 1]]
            + smWgt[pk * 4 + 2] * basep[smOfs[pk * 4 + 2]]
            + smWgt[pk * 4 + 3] * basep[smOfs[pk * 4 + 3]];
    smAe[frag_swizzle(kd, p)] = (__bf16)v;
  }
  __syncthreads();

  // ---- Phase 3: main einsum, 36 x wmma per wave (8 N-tiles = 128 ch) ----
  const int n = wave * 16 + ml;             // global output channel
  const v16bf* bbase = pw + ((size_t)n * 2 + half);
  v8f acc = {};
  for (int kc = 0; kc < NKC; ++kc) {
    v16bf a  = smA[(kc * 16 + ml) * 2 + half];
    v16bf bf = bbase[(size_t)kc * 256];     // 128 n * 2 half frags per chunk
    if (kc + 1 < NKC)
      __builtin_prefetch(&bbase[(size_t)(kc + 1) * 256], 0, 3);
    acc = __builtin_amdgcn_wmma_f32_16x16x32_bf16(
        false, a, false, bf, (short)0, acc, false, false);
  }

  // D layout: VGPR r -> row m = r + 8*half, col n (bias cancels in inorm)
  float* obase = out + ((size_t)(b * Cc + n) * Hh + y) * Ww + x0 + 8 * half;
  #pragma unroll
  for (int r = 0; r < 8; ++r) obase[r] = acc[r];
}

// ---------------------------------------------------------------------------
// Instance norm per (b,c) plane + optional relu + optional residual.
// Conv bias is a per-channel constant -> cancelled by mean subtraction.
// ---------------------------------------------------------------------------
__global__ __launch_bounds__(256) void inorm_kernel(
    const float* __restrict__ g, const float* __restrict__ resid,
    float* __restrict__ out, int do_relu) {
  __shared__ float ssum[256], ssq[256];
  int plane = blockIdx.x;                   // b*C + c
  const float* gp = g + (size_t)plane * HW;
  float s = 0.f, q = 0.f;
  for (int i = threadIdx.x; i < HW; i += 256) {
    float v = gp[i];
    s += v;
    q = fmaf(v, v, q);
  }
  ssum[threadIdx.x] = s;
  ssq[threadIdx.x]  = q;
  __syncthreads();
  for (int st = 128; st > 0; st >>= 1) {
    if (threadIdx.x < st) {
      ssum[threadIdx.x] += ssum[threadIdx.x + st];
      ssq[threadIdx.x]  += ssq[threadIdx.x + st];
    }
    __syncthreads();
  }
  float mean = ssum[0] * (1.0f / (float)HW);
  float var  = ssq[0] * (1.0f / (float)HW) - mean * mean;
  float inv  = rsqrtf(var + 1e-5f);
  float* op = out + (size_t)plane * HW;
  const float* rp = resid ? resid + (size_t)plane * HW : nullptr;
  for (int i = threadIdx.x; i < HW; i += 256) {
    float v = (gp[i] - mean) * inv;
    if (do_relu) v = fmaxf(v, 0.f);
    if (rp) v += rp[i];
    op[i] = v;
  }
}

// ---------------------------------------------------------------------------
// Orchestration.  Workspace (~38.5 MB):
//   pw1[294912] | pw2[294912] | pwom1[73728] | pwom2[73728] | g[37748736]
// r1 is staged in d_out (dead before the final inorm overwrites it).
// ---------------------------------------------------------------------------
extern "C" void kernel_launch(void* const* d_in, const int* in_sizes, int n_in,
                              void* d_out, int out_size, void* d_ws,
                              size_t ws_size, hipStream_t stream) {
  (void)in_sizes; (void)n_in; (void)out_size; (void)ws_size;
  const float* x    = (const float*)d_in[0];
  const float* lo   = (const float*)d_in[1];
  const float* wom1 = (const float*)d_in[2];
  const float* bom1 = (const float*)d_in[3];
  const float* w1   = (const float*)d_in[4];
  // d_in[5] = b1: cancels under instance norm
  const float* wom2 = (const float*)d_in[6];
  const float* bom2 = (const float*)d_in[7];
  const float* w2   = (const float*)d_in[8];
  // d_in[9] = b2: cancels under instance norm
  float* out = (float*)d_out;

  char* ws = (char*)d_ws;
  __bf16* pw1   = (__bf16*)ws;
  __bf16* pw2   = (__bf16*)(ws + 294912);
  __bf16* pwom1 = (__bf16*)(ws + 2 * 294912);
  __bf16* pwom2 = (__bf16*)(ws + 2 * 294912 + 73728);
  float*  g     = (float*)(ws + 2 * 294912 + 2 * 73728);

  pack_w_kernel<<<576, 256, 0, stream>>>(w1, pw1);
  pack_w_kernel<<<576, 256, 0, stream>>>(w2, pw2);
  pack_wom_kernel<<<144, 256, 0, stream>>>(wom1, pwom1);
  pack_wom_kernel<<<144, 256, 0, stream>>>(wom2, pwom2);

  // Layer 1: fused dcn(x, lo) -> g ; relu(inorm(g)) -> r1 (in d_out)
  dcn_fused_kernel<<<4608, 256, 0, stream>>>(x, lo, (const v16bf*)pwom1, bom1,
                                             (const v16bf*)pw1, g);
  inorm_kernel<<<256, 256, 0, stream>>>(g, nullptr, out, 1);

  // Layer 2: fused dcn(r1, lo) -> g ; x + inorm(g) -> out
  dcn_fused_kernel<<<4608, 256, 0, stream>>>(out, lo, (const v16bf*)pwom2, bom2,
                                             (const v16bf*)pw2, g);
  inorm_kernel<<<256, 256, 0, stream>>>(g, x, out, 0);
}
